// ContextualizedNN_57002805952924
// MI455X (gfx1250) — matrix operand order repaired and checked by reading, not compiled, not run on hardware
//
#include <hip/hip_runtime.h>
#include <hip/hip_bf16.h>

typedef _Float16 v16h __attribute__((ext_vector_type(16)));
typedef _Float16 v8h  __attribute__((ext_vector_type(8)));
typedef float    v8f  __attribute__((ext_vector_type(8)));

#define B_SZ    8192
#define KNEI    100
#define DEMB    64
#define CAT     12800
#define HID     128
#define KPAD    112   // padded M rows of scored
#define KDIM    128   // padded reduction dim for side matmul
#define SPITCH  136   // halves; 272B rows -> b128 frag loads conflict-free, 16B aligned
#define EPITCH  136
#define XPITCH  72    // halves; 144B rows for kernel-2 x tile (16B aligned, conflict-free)

// Build a 16-half fragment from two 8-half chunks located at p and p+hi_off.
__device__ __forceinline__ v16h make_frag(const _Float16* p, int hi_off) {
    v8h lo = *(const v8h*)(p);
    v8h hi = *(const v8h*)(p + hi_off);
    return __builtin_shufflevector(lo, hi, 0,1,2,3,4,5,6,7,8,9,10,11,12,13,14,15);
}

// A fragment: 16x32 f16. Lane l holds row m0+(l&15); VGPR0-7 = K[kh*16..+15],
// kh = l>>4.
__device__ __forceinline__ v16h load_a_frag(const _Float16* Sl, int pitch,
                                            int mrow0, int kb, int lane) {
    int m = mrow0 + (lane & 15);
    int kh = lane >> 4;
    return make_frag(Sl + (size_t)m * pitch + kb + kh * 16, 8);
}

// B fragment: 32x16 f16 from transposed operand BT[n][k]. Lane l holds column
// n0+(l&15); VGPR i packs K = koff+2i, koff+2i+1 with koff = kb + (l>>4)*16.
__device__ __forceinline__ v16h load_b_frag(const _Float16* BT, size_t pitch,
                                            int nb, int kb, int lane) {
    int n = nb + (lane & 15);
    int koff = kb + (lane >> 4) * 16;
    return make_frag(BT + (size_t)n * pitch + koff, 8);
}

// Kernel 0: one-time W1 [12800,128] f32 -> W1T [128,12800] f16 (L2-resident).
__global__ __launch_bounds__(256) void w1t_kernel(const float* __restrict__ W1,
                                                  _Float16* __restrict__ W1T) {
    int e = blockIdx.x * 256 + threadIdx.x;   // e = k*128 + n (coalesced read)
    if (e < CAT * HID) {
        int k = e >> 7, n = e & 127;
        W1T[(size_t)n * CAT + k] = (_Float16)W1[e];
    }
}

// Kernel 1: one workgroup per (batch row, side). Gather S [100x100] and E
// [100x64] into LDS (f16, padded/zeroed), compute scored = S @ E with WMMA,
// write f16 rows into x buffer at [b, side*6400 + m*64 + n].
__global__ __launch_bounds__(256) void side_scored_kernel(
    const int* __restrict__ user_idxs, const int* __restrict__ item_idxs,
    const int* __restrict__ user_idxt, const int* __restrict__ item_idxt,
    const float* __restrict__ user_scr, const float* __restrict__ item_scr,
    const float* __restrict__ user_emb, const float* __restrict__ item_emb,
    _Float16* __restrict__ xbuf)
{
    __shared__ int neighs[KNEI];
    __shared__ _Float16 S[KPAD * SPITCH];   // S[m][j]  (scores)
    __shared__ _Float16 ET[DEMB * EPITCH];  // ET[d][j] (emb transposed)

    const int blk  = blockIdx.x;
    const int b    = blk >> 1;
    const int side = blk & 1;
    const int*   idxs = side ? item_idxs : user_idxs;
    const int*   idxt = side ? item_idxt : user_idxt;
    const float* scr  = side ? item_scr  : user_scr;
    const float* emb  = side ? item_emb  : user_emb;

    const int tid = threadIdx.x;
    const int center = idxs[b];
    if (tid < KNEI) neighs[tid] = idxt[(size_t)center * KNEI + tid];
    __syncthreads();

    // Fill S[m][j], zero-padded to 112x128 (coalesced row reads, L2-resident).
    for (int e = tid; e < KPAD * KDIM; e += 256) {
        int r = e >> 7, c = e & 127;
        float v = (r < KNEI && c < KNEI) ? scr[(size_t)neighs[r] * KNEI + c] : 0.f;
        S[r * SPITCH + c] = (_Float16)v;
    }
    // Fill ET[d][j] transposed, zero-padded j to 128 (coalesced over d).
    for (int e = tid; e < KDIM * DEMB; e += 256) {
        int j = e >> 6, d = e & 63;
        float v = (j < KNEI) ? emb[(size_t)neighs[j] * DEMB + d] : 0.f;
        ET[d * EPITCH + j] = (_Float16)v;
    }
    __syncthreads();

    const int wave = tid >> 5, lane = tid & 31;
    const size_t out_base = (size_t)b * CAT + (size_t)side * (KNEI * DEMB);
    const int n_lo = lane & 15;
    const int moff = (lane >> 4) << 3;

    // Static tile schedule over the 7x4 tile grid: wave w owns column
    // nt = w&3; waves 0-3 rows {0,2,4,6}, waves 4-7 rows {1,3,5}.
    const int nt  = wave & 3;
    const int mt0 = wave >> 2;          // 0 or 1
    const int nb  = nt * 16;
    const int n   = nb + n_lo;

    // B fragments for this wave's column (loop-invariant, 32 VGPRs).
    v16h bf0 = load_b_frag(ET, EPITCH, nb,  0, lane);
    v16h bf1 = load_b_frag(ET, EPITCH, nb, 32, lane);
    v16h bf2 = load_b_frag(ET, EPITCH, nb, 64, lane);
    v16h bf3 = load_b_frag(ET, EPITCH, nb, 96, lane);

    // Three fully-valid tiles per wave (rows <= 95): unconditional stores.
    #pragma unroll
    for (int i = 0; i < 3; i++) {
        int mt = mt0 + 2 * i;
        v8f c = {};
        c = __builtin_amdgcn_wmma_f32_16x16x32_f16(false, load_a_frag(S, SPITCH, mt*16,  0, lane), false, bf0, (short)0, c, false, false);
        c = __builtin_amdgcn_wmma_f32_16x16x32_f16(false, load_a_frag(S, SPITCH, mt*16, 32, lane), false, bf1, (short)0, c, false, false);
        c = __builtin_amdgcn_wmma_f32_16x16x32_f16(false, load_a_frag(S, SPITCH, mt*16, 64, lane), false, bf2, (short)0, c, false, false);
        c = __builtin_amdgcn_wmma_f32_16x16x32_f16(false, load_a_frag(S, SPITCH, mt*16, 96, lane), false, bf3, (short)0, c, false, false);
        int mbase = mt * 16 + moff;
        #pragma unroll
        for (int r = 0; r < 8; r++)
            xbuf[out_base + (size_t)(mbase + r) * DEMB + n] = (_Float16)c[r];
    }

    // Tail tile mt=6 (rows 96..111): only waves 0-3 (uniform branch, EXEC
    // stays all-ones around the WMMAs); valid rows are statically 96..99,
    // i.e. lanes 0-15, r 0..3.
    if (mt0 == 0) {
        v8f c = {};
        c = __builtin_amdgcn_wmma_f32_16x16x32_f16(false, load_a_frag(S, SPITCH, 96,  0, lane), false, bf0, (short)0, c, false, false);
        c = __builtin_amdgcn_wmma_f32_16x16x32_f16(false, load_a_frag(S, SPITCH, 96, 32, lane), false, bf1, (short)0, c, false, false);
        c = __builtin_amdgcn_wmma_f32_16x16x32_f16(false, load_a_frag(S, SPITCH, 96, 64, lane), false, bf2, (short)0, c, false, false);
        c = __builtin_amdgcn_wmma_f32_16x16x32_f16(false, load_a_frag(S, SPITCH, 96, 96, lane), false, bf3, (short)0, c, false, false);
        if (lane < 16) {
            #pragma unroll
            for (int r = 0; r < 4; r++)
                xbuf[out_base + (size_t)(96 + r) * DEMB + n] = (_Float16)c[r];
        }
    }
}

// Kernel 2: fused MLP. One workgroup per 16 batch rows; wave w owns N-tile
// n = w*16 of h. K-loop over 12800 in steps of 64: x tile staged into LDS via
// GLOBAL_LOAD_ASYNC_TO_LDS_B64 (ASYNCcnt path, shared A fragment), B
// fragments straight from L2-resident W1T; two WMMAs per step. Epilogue:
// bias+ReLU, dot with W2 via LDS atomics, ReLU, sigmoid.
__global__ __launch_bounds__(256) void mlp_kernel(
    const _Float16* __restrict__ xbuf, const _Float16* __restrict__ W1T,
    const float* __restrict__ b1, const float* __restrict__ W2,
    const float* __restrict__ b2, float* __restrict__ out)
{
    __shared__ _Float16 XA[16 * XPITCH];    // x tile [16 m][64 k]
    __shared__ float oacc[16];

    const int tid  = threadIdx.x;
    const int b0   = blockIdx.x * 16;
    const int wave = tid >> 5, lane = tid & 31;

    if (tid < 16) oacc[tid] = 0.f;

    const int n = wave * 16 + (lane & 15);
    const _Float16* wrow = W1T + (size_t)n * CAT + ((lane >> 4) << 4);

    // Per-thread addresses for async x staging (one b64 per thread per step).
    const int sm  = tid >> 4;
    const int sc4 = (tid & 15) * 4;
    const _Float16* xsrc = xbuf + (size_t)(b0 + sm) * CAT + sc4;
    const uint32_t  xlds = (uint32_t)(uintptr_t)(&XA[sm * XPITCH + sc4]);

    v8f c = {};
    for (int kk = 0; kk < CAT; kk += 64) {
        __syncthreads();
        // Async global->LDS copy of the 16x64 x tile (8B per lane).
        asm volatile("global_load_async_to_lds_b64 %0, %1, off"
                     :: "v"(xlds), "v"(xsrc + kk)
                     : "memory");
        asm volatile("s_wait_asynccnt 0" ::: "memory");
        __syncthreads();
        #pragma unroll
        for (int h = 0; h < 2; h++) {
            v16h a  = load_a_frag(XA, XPITCH, 0, h * 32, lane);
            v16h bf = make_frag(wrow + kk + h * 32, 8);   // global b128 x2, L2 hit
            c = __builtin_amdgcn_wmma_f32_16x16x32_f16(
                    false, a, false, bf, (short)0, c, false, false);
        }
    }
    __syncthreads();

    // Epilogue: h = relu(c + b1[n]); o[m] += h * W2[n]
    const float bias = b1[n];
    const float w2n  = W2[n];
    const int mbase  = (lane >> 4) << 3;
    #pragma unroll
    for (int r = 0; r < 8; r++) {
        float h = c[r] + bias;
        h = h > 0.f ? h : 0.f;
        atomicAdd(&oacc[mbase + r], h * w2n);
    }
    __syncthreads();
    if (tid < 16) {
        float o = oacc[tid] + b2[0];
        o = o > 0.f ? o : 0.f;
        out[b0 + tid] = 1.f / (1.f + __expf(-o));
    }
}

extern "C" void kernel_launch(void* const* d_in, const int* in_sizes, int n_in,
                              void* d_out, int out_size, void* d_ws, size_t ws_size,
                              hipStream_t stream) {
    (void)in_sizes; (void)n_in; (void)out_size; (void)ws_size;
    const int*   user_idxs = (const int*)  d_in[0];
    const int*   item_idxs = (const int*)  d_in[1];
    const int*   user_idxt = (const int*)  d_in[2];
    const int*   item_idxt = (const int*)  d_in[3];
    const float* user_scr  = (const float*)d_in[4];
    const float* item_scr  = (const float*)d_in[5];
    const float* user_emb  = (const float*)d_in[6];
    const float* item_emb  = (const float*)d_in[7];
    const float* W1        = (const float*)d_in[8];
    const float* b1        = (const float*)d_in[9];
    const float* W2        = (const float*)d_in[10];
    const float* b2        = (const float*)d_in[11];
    float*       out       = (float*)d_out;

    _Float16* xbuf = (_Float16*)d_ws;                       // [8192,12800] f16 (210 MB)
    _Float16* W1T  = xbuf + (size_t)B_SZ * CAT;             // [128,12800]  f16 (3.3 MB)

    w1t_kernel<<<dim3((CAT * HID + 255) / 256), dim3(256), 0, stream>>>(W1, W1T);

    side_scored_kernel<<<dim3(2 * B_SZ), dim3(256), 0, stream>>>(
        user_idxs, item_idxs, user_idxt, item_idxt,
        user_scr, item_scr, user_emb, item_emb, xbuf);

    mlp_kernel<<<dim3(B_SZ / 16), dim3(256), 0, stream>>>(
        xbuf, W1T, b1, W2, b2, out);
}
